// RelPositionMultiheadAttentionWeights_67688684585481
// MI455X (gfx1250) — compile-verified, matrix operand order checked by reading, not run on hardware
//
#include <hip/hip_runtime.h>
#include <hip/hip_bf16.h>

typedef __attribute__((ext_vector_type(16))) _Float16     v16h;
typedef __attribute__((ext_vector_type(8)))  float        v8f;
typedef __attribute__((ext_vector_type(4)))  unsigned int u32x4;
typedef __attribute__((ext_vector_type(8)))  int          i32x8;
typedef __attribute__((ext_vector_type(4)))  int          i32x4;

#define NHEADS 5
#define QHD    32
#define PHD    4
#define BDIM   256
#define SLEN   256
#define EMB    512
#define POSD   192
#define RELN   (2*BDIM-1)   /* 511 */
#define TM     32           /* rows per attention block */

// Convert 16 consecutive fp32 (4x float4, global or LDS) to a v16h f16 vector.
__device__ __forceinline__ v16h cvt16(const float4* p) {
    v16h r;
#pragma unroll
    for (int i = 0; i < 4; ++i) {
        float4 t = p[i];
        r[4*i+0] = (_Float16)t.x;
        r[4*i+1] = (_Float16)t.y;
        r[4*i+2] = (_Float16)t.z;
        r[4*i+3] = (_Float16)t.w;
    }
    return r;
}

// Issue a TDM 2D tile load: W[0..tile_rows)[kcol0..kcol0+32) (fp32) -> LDS slab.
// Rows >= n_valid are OOB and return zero (TDM zero-fills), giving free padding.
__device__ __forceinline__ void tdm_load_wslab(const float* W, int kcol0,
                                               unsigned lds_base, int n_valid,
                                               int tile_rows) {
    unsigned long long ga = (unsigned long long)(const void*)(W + kcol0);
    u32x4 g0;
    g0[0] = 1u;                                              // count=1, no gather
    g0[1] = lds_base;                                        // LDS byte address
    g0[2] = (unsigned)ga;                                    // global addr lo
    g0[3] = (unsigned)((ga >> 32) & 0x1FFFFFFull) | (2u << 30); // addr hi | type=2
    i32x8 g1;
    g1[0] = (int)(2u << 16);                                 // data_size = 4 bytes
    g1[1] = (int)(32u << 16);                                // tensor_dim0 = 32 (lo16)
    g1[2] = (int)(((unsigned)n_valid & 0xFFFFu) << 16);      // dim0 hi=0 | tensor_dim1 lo
    g1[3] = (int)(32u << 16);                                // tensor_dim1 hi=0 | tile_dim0=32
    g1[4] = (int)(unsigned)tile_rows;                        // tile_dim1 | tile_dim2=0
    g1[5] = (int)EMB;                                        // tensor_dim0_stride = 512
    g1[6] = 0;                                               // stride hi | dim1_stride lo
    g1[7] = 0;
    i32x4 z4 = {};
#if defined(__clang_major__) && __clang_major__ >= 23
    i32x8 z8 = {};
    __builtin_amdgcn_tensor_load_to_lds(g0, g1, z4, z4, z8, 0);
#else
    __builtin_amdgcn_tensor_load_to_lds(g0, g1, z4, z4, 0);
#endif
}

// ---------------------------------------------------------------------------
// Projection GEMM:  Y = X @ W^T + bias,  X: (B*S, 512), W: (N, 512)
// Block = 4 waves; wave w owns M-tile (blockIdx.x*4 + w) and keeps ALL NT
// 16x16 N-tile accumulators in registers.  Per K-step (K=512 = 16*32):
//   - wave0 TDM-loads the shared W K-slab (NT*16 x 32 fp32) into LDS,
//     s_wait_tensorcnt, barrier
//   - each wave loads its A tile once (fp32->f16) and runs NT WMMAs from LDS
// Output scattered into attention layout:
//   n <  160           -> qk_out[h][s][b][d] (f16)
//   160 <= n < N(=180) -> p_out [h][s][b][d] (fp32)
// ---------------------------------------------------------------------------
template<int NT>
__global__ __launch_bounds__(128)
void proj_kernel(const float* __restrict__ X, const float* __restrict__ W,
                 const float* __restrict__ bias, int N,
                 _Float16* __restrict__ qk_out, float* __restrict__ p_out)
{
    __shared__ float wslab[192 * 32];                 // 24 KB, [n][kcol]

    const int tid    = threadIdx.x;
    const int wave   = tid >> 5;
    const int l      = tid & 31;
    const int lane16 = l & 15;
    const int half   = l >> 4;
    const int m0     = (blockIdx.x * 4 + wave) * 16;  // 16 rows over (b*S + s)
    const int b      = m0 / SLEN;                     // 16 | 256 -> same b per tile
    const int s0     = m0 % SLEN;

    const float4* xv = (const float4*)(X + (size_t)b * SLEN * EMB
                                         + (size_t)(s0 + lane16) * EMB + half * 16);

    v8f C[NT];
#pragma unroll
    for (int j = 0; j < NT; ++j) C[j] = (v8f){};

    const unsigned lds_base = (unsigned)(size_t)(void*)wslab;

#pragma unroll 1
    for (int kk = 0; kk < 16; ++kk) {
        __syncthreads();                              // LDS slab free for reuse
        if (wave == 0) {
            tdm_load_wslab(W, kk * 32, lds_base, N, NT * 16);
            __builtin_amdgcn_s_wait_tensorcnt(0);     // DMA complete for wave0
        }
        __syncthreads();                              // slab published to block

        if (kk + 2 < 16) __builtin_prefetch(xv + (kk + 2) * 8, 0, 1);
        v16h a = cvt16(xv + kk * 8);                  // A: lane=row, 16 K elems

#pragma unroll
        for (int nj = 0; nj < NT; ++nj) {             // NT back-to-back WMMAs
            const float4* bsrc =
                (const float4*)&wslab[(nj * 16 + lane16) * 32 + half * 16];
            v16h bm = cvt16(bsrc);                    // B: lane=col (W row), 16 K
            C[nj] = __builtin_amdgcn_wmma_f32_16x16x32_f16(false, a, false, bm,
                                                           (short)0, C[nj],
                                                           false, false);
        }
    }

    // C layout: VGPR r -> M = r + 8*half ; N = lane&15
#pragma unroll
    for (int nj = 0; nj < NT; ++nj) {
        int n = nj * 16 + lane16;
        if (n < N) {
            float bv = bias[n];
#pragma unroll
            for (int r = 0; r < 8; ++r) {
                int   s   = s0 + r + half * 8;
                float val = C[nj][r] + bv;
                if (n < NHEADS * QHD) {
                    int hh = n / QHD, d = n % QHD;
                    qk_out[(((size_t)hh * SLEN + s) * BDIM + b) * QHD + d] =
                        (_Float16)val;
                } else if (p_out) {
                    int n2 = n - NHEADS * QHD;
                    int hh = n2 / PHD, d = n2 % PHD;
                    p_out[(((size_t)hh * SLEN + s) * BDIM + b) * PHD + d] = val;
                }
            }
        }
    }
}

// ---------------------------------------------------------------------------
// pe[h][d][rel] = sum_c pos_emb[rel][c] * W_pos[h*4+d][c]     (tiny GEMM)
// ---------------------------------------------------------------------------
__global__ void pe_kernel(const float* __restrict__ pos_emb,
                          const float* __restrict__ W_pos,
                          float* __restrict__ pe)
{
    int idx = blockIdx.x * blockDim.x + threadIdx.x;   // over 20 * 511
    if (idx >= NHEADS * PHD * RELN) return;
    int r  = idx % RELN;
    int hd = idx / RELN;
    const float* pr = pos_emb + (size_t)r * POSD;
    const float* wr = W_pos   + (size_t)hd * POSD;
    float acc = 0.f;
#pragma unroll 4
    for (int c = 0; c < POSD; ++c) acc += pr[c] * wr[c];
    pe[idx] = acc;                                     // layout [hd][rel]
}

// ---------------------------------------------------------------------------
// Attention: one block per (h,s, 32-row strip).
//   scores[i][j] = sum_d q[h,s,i,d] * kT[h,s,j,d]          (WMMA, K=32)
//   scores[i][j] += sum_d p[h,s,i,d] * pe[h,d, 255-i+j]    (rel-shift fused)
//   out = softmax_j(scores)
// ---------------------------------------------------------------------------
__global__ __launch_bounds__(256)
void attn_kernel(const _Float16* __restrict__ q, const _Float16* __restrict__ kT,
                 const float* __restrict__ p, const float* __restrict__ pe,
                 float* __restrict__ out)
{
    __shared__ float sc[TM * BDIM];                    // 32 KB

    const int hs      = blockIdx.x;                    // 0..1279
    const int h       = hs / SLEN;
    const int s       = hs % SLEN;
    const int rowbase = blockIdx.y * TM;
    const int tid     = threadIdx.x;
    const int wave    = tid >> 5;
    const int l       = tid & 31;
    const int lane16  = l & 15;
    const int half    = l >> 4;

    const _Float16* qbase = q  + ((size_t)h * SLEN + s) * BDIM * QHD;
    const _Float16* kbase = kT + ((size_t)h * SLEN + s) * BDIM * QHD;

    // 2 (M) x 16 (N) = 32 tiles of 16x16; 4 tiles per wave; K=32 -> 1 WMMA each
    for (int t = wave; t < 32; t += 8) {
        int mi = t >> 4;
        int nj = t & 15;
        v16h a  = *(const v16h*)(qbase + (size_t)(rowbase + mi * 16 + lane16) * QHD + half * 16);
        v16h bm = *(const v16h*)(kbase + (size_t)(nj * 16 + lane16) * QHD + half * 16);
        v8f  c  = {};
        c = __builtin_amdgcn_wmma_f32_16x16x32_f16(false, a, false, bm,
                                                   (short)0, c, false, false);
#pragma unroll
        for (int r = 0; r < 8; ++r) {
            int mloc = mi * 16 + r + half * 8;
            int col  = nj * 16 + lane16;
            sc[mloc * BDIM + col] = c[r];
        }
    }
    __syncthreads();

    // fused relative-shift positional scores: rel = (B-1) - i + j in [0, 510]
    const float* pev = pe + (size_t)h * PHD * RELN;
    for (int idx = tid; idx < TM * BDIM; idx += 256) {
        int iloc = idx >> 8;
        int j    = idx & 255;
        int i    = rowbase + iloc;
        int rel  = (BDIM - 1) - i + j;
        const float* pv = p + (((size_t)h * SLEN + s) * BDIM + i) * PHD;
        float acc = pv[0] * pev[rel]
                  + pv[1] * pev[RELN + rel]
                  + pv[2] * pev[2 * RELN + rel]
                  + pv[3] * pev[3 * RELN + rel];
        sc[idx] += acc;
    }
    __syncthreads();

    // row softmax: wave w handles rows w, w+8, w+16, w+24 (wave32 shuffles)
    for (int rr = wave; rr < TM; rr += 8) {
        float v[8];
        float mx = -3.0e38f;
#pragma unroll
        for (int jj = 0; jj < 8; ++jj) {
            v[jj] = sc[rr * BDIM + jj * 32 + l];
            mx = fmaxf(mx, v[jj]);
        }
#pragma unroll
        for (int off = 16; off >= 1; off >>= 1) mx = fmaxf(mx, __shfl_xor(mx, off, 32));
        float sum = 0.f;
#pragma unroll
        for (int jj = 0; jj < 8; ++jj) { v[jj] = __expf(v[jj] - mx); sum += v[jj]; }
#pragma unroll
        for (int off = 16; off >= 1; off >>= 1) sum += __shfl_xor(sum, off, 32);
        float inv = 1.0f / sum;
        int i = rowbase + rr;
        float* orow = out + (((size_t)h * SLEN + s) * BDIM + i) * BDIM;
#pragma unroll
        for (int jj = 0; jj < 8; ++jj) orow[jj * 32 + l] = v[jj] * inv;
    }
}

// ---------------------------------------------------------------------------
extern "C" void kernel_launch(void* const* d_in, const int* in_sizes, int n_in,
                              void* d_out, int out_size, void* d_ws, size_t ws_size,
                              hipStream_t stream)
{
    const float* lm_pruned = (const float*)d_in[0];
    const float* am_pruned = (const float*)d_in[1];
    const float* pos_emb   = (const float*)d_in[2];
    const float* W_lm      = (const float*)d_in[3];
    const float* b_lm      = (const float*)d_in[4];
    const float* W_am      = (const float*)d_in[5];
    const float* b_am      = (const float*)d_in[6];
    const float* W_pos     = (const float*)d_in[7];
    float* out = (float*)d_out;

    // workspace layout (all fully rewritten each launch -> deterministic)
    char* ws = (char*)d_ws;
    _Float16* q  = (_Float16*)(ws);                       // 20,971,520 B
    _Float16* kT = (_Float16*)(ws + 20971520);            // 20,971,520 B
    float*    p  = (float*)   (ws + 41943040);            //  5,242,880 B
    float*    pe = (float*)   (ws + 47185920);            //     40,880 B

    // lm projection: N=180 -> q (f16) + p (fp32); 4096 M-tiles / 4 waves
    proj_kernel<12><<<dim3(1024), 128, 0, stream>>>(lm_pruned, W_lm, b_lm, 180, q, p);
    // am projection: N=160 -> kT (f16)
    proj_kernel<10><<<dim3(1024), 128, 0, stream>>>(am_pruned, W_am, b_am, 160, kT, nullptr);
    // positional projection (tiny)
    pe_kernel<<<(NHEADS * PHD * RELN + 255) / 256, 256, 0, stream>>>(pos_emb, W_pos, pe);
    // attention scores + rel-shift + softmax
    attn_kernel<<<dim3(NHEADS * SLEN, BDIM / TM), 256, 0, stream>>>(q, kT, p, pe, out);
}
